// MultiHeadAttention_33423435498059
// MI455X (gfx1250) — matrix operand order repaired and checked by reading, not compiled, not run on hardware
//
#include <hip/hip_runtime.h>
#include <hip/hip_bf16.h>

typedef __bf16 bf16;
typedef __attribute__((ext_vector_type(4)))  __bf16 bf16x4;
typedef __attribute__((ext_vector_type(8)))  __bf16 bf16x8;
typedef __attribute__((ext_vector_type(16))) __bf16 bf16x16;
typedef __attribute__((ext_vector_type(8)))  float   f32x8;

// ---------------------------------------------------------------------------
// WMMA helpers (CDNA5: D = A(16x32 bf16) x B(32x16 bf16) + C(16x16 f32))
// ---------------------------------------------------------------------------
__device__ __forceinline__ f32x8 wmma_bf16(bf16x16 a, bf16x16 b, f32x8 c) {
  return __builtin_amdgcn_wmma_f32_16x16x32_bf16(
      /*neg_a=*/false, a, /*neg_b=*/false, b,
      /*c_mod=*/(short)0, c, /*reuse_a=*/false, /*reuse_b=*/false);
}

// A fragment from a row-major [16 x K] LDS tile.
// lanes 0-15: row=lane,   K {0..7} then {16..23} (relative to koff)
// lanes 16-31: row=lane-16, K {8..15} then {24..31}
__device__ __forceinline__ bf16x16 frag_a(const bf16* base, int stride, int koff) {
  const int lane = threadIdx.x & 31;
  const int row = lane & 15, half = lane >> 4;
  const bf16* p = base + row * stride + koff + half * 8;
  bf16x8 lo = *(const bf16x8*)p;
  bf16x8 hi = *(const bf16x8*)(p + 16);
  bf16x16 r;
#pragma unroll
  for (int i = 0; i < 8; ++i) { r[i] = lo[i]; r[i + 8] = hi[i]; }
  return r;
}

// B fragment from an LDS tile stored transposed as [N=16 rows][K] (i.e. B^T).
// lanes 0-15: col=lane, K {0..7} then {16..23}; lanes 16-31: K {8..15} then {24..31}.
__device__ __forceinline__ bf16x16 frag_b(const bf16* base, int stride, int koff) {
  const int lane = threadIdx.x & 31;
  const int col = lane & 15, half = lane >> 4;
  const bf16* p = base + col * stride + koff + half * 8;
  bf16x8 lo = *(const bf16x8*)p;
  bf16x8 hi = *(const bf16x8*)(p + 16);
  bf16x16 r;
#pragma unroll
  for (int i = 0; i < 8; ++i) { r[i] = lo[i]; r[i + 8] = hi[i]; }
  return r;
}

// ---------------------------------------------------------------------------
// fp32 -> bf16 conversion (x), vectorized x4
// ---------------------------------------------------------------------------
__global__ __launch_bounds__(256) void cvt_f32_bf16(const float* __restrict__ in,
                                                    bf16* __restrict__ out, int n4) {
  int i = blockIdx.x * blockDim.x + threadIdx.x;
  if (i >= n4) return;
  const float4 f = ((const float4*)in)[i];
  bf16x4 o;
  o[0] = (bf16)f.x; o[1] = (bf16)f.y; o[2] = (bf16)f.z; o[3] = (bf16)f.w;
  ((bf16x4*)out)[i] = o;
}

// fused convert + transpose of weights: WT[n*1024 + k] = (bf16)W[k*1024 + n]
// one-time pass; reads coalesced along n.
__global__ __launch_bounds__(256) void cvt_w_transpose(const float* __restrict__ W,
                                                       bf16* __restrict__ WT) {
  const int idx = blockIdx.x * 256 + threadIdx.x;  // 1M threads
  const int n = idx & 1023, k = idx >> 10;
  WT[(size_t)n * 1024 + k] = (bf16)W[(size_t)k * 1024 + n];
}

// tiled transpose of V: VT[((b*16+h)*64 + d)*1024 + l] = V[((b*1024+l)*16+h)*64 + d]
__global__ __launch_bounds__(256) void transpose_v(const bf16* __restrict__ V,
                                                   bf16* __restrict__ VT) {
  __shared__ __align__(16) bf16 t[64 * 72];
  const int bh = blockIdx.y;
  const int b = bh >> 4, h = bh & 15;
  const int l0 = blockIdx.x * 64;
  const int tid = threadIdx.x;
  {
    const int r = tid >> 2, seg = tid & 3;  // r: l-local row, seg*16 in d
    const bf16* g = V + ((size_t)((b * 1024 + l0 + r) * 16 + h)) * 64 + seg * 16;
    *(bf16x8*)&t[r * 72 + seg * 16]     = *(const bf16x8*)g;
    *(bf16x8*)&t[r * 72 + seg * 16 + 8] = *(const bf16x8*)(g + 8);
  }
  __syncthreads();
  {
    const int d = tid >> 2, seg = tid & 3;  // seg*16 in l
    bf16x8 o0, o1;
#pragma unroll
    for (int j = 0; j < 8; ++j) {
      o0[j] = t[(seg * 16 + j) * 72 + d];
      o1[j] = t[(seg * 16 + 8 + j) * 72 + d];
    }
    bf16* g = VT + ((size_t)(bh * 64 + d)) * 1024 + l0 + seg * 16;
    *(bf16x8*)g = o0;
    *(bf16x8*)(g + 8) = o1;
  }
}

// ---------------------------------------------------------------------------
// GEMM: O[8192,1024] = X[8192,1024] * W + bias, W given pre-transposed WT[n][k].
// 128x64 tile/WG, k-slice 64, double-buffered LDS, register-staged pipeline,
// ONE barrier per k-iteration. 8 waves; each wave: 16-row strip x 64 cols.
// ---------------------------------------------------------------------------
__global__ __launch_bounds__(256) void gemm_qkv(const bf16* __restrict__ X,
                                                const bf16* __restrict__ WT,
                                                const float* __restrict__ bias,
                                                bf16* __restrict__ O) {
  __shared__ __align__(16) bf16 Xs[2][128 * 72];  // [row][k], stride 72 (144B)
  __shared__ __align__(16) bf16 Ws[2][64 * 72];   // B^T tile: [n][k]
  const int m0 = blockIdx.y * 128;
  const int n0 = blockIdx.x * 64;
  const int tid = threadIdx.x;
  const int w = tid >> 5, lane = tid & 31;
  const int col = lane & 15, half = lane >> 4;

  // staging maps: X 32 bf16/thread, W 16 bf16/thread (all b128)
  const int xrow = tid >> 1, xh = tid & 1;
  const int wn = tid >> 2, wseg = tid & 3;
  const bf16* xg = X + (size_t)(m0 + xrow) * 1024 + xh * 32;
  const bf16* wg = WT + (size_t)(n0 + wn) * 1024 + wseg * 16;

  f32x8 acc[4];
#pragma unroll
  for (int t = 0; t < 4; ++t) { f32x8 z = {}; acc[t] = z; }

  bf16x8 xr[4], wr[2];
  // prologue: stage k-slice 0
#pragma unroll
  for (int j = 0; j < 4; ++j) xr[j] = *(const bf16x8*)(xg + j * 8);
  wr[0] = *(const bf16x8*)wg;
  wr[1] = *(const bf16x8*)(wg + 8);
#pragma unroll
  for (int j = 0; j < 4; ++j) *(bf16x8*)&Xs[0][xrow * 72 + xh * 32 + j * 8] = xr[j];
  *(bf16x8*)&Ws[0][wn * 72 + wseg * 16] = wr[0];
  *(bf16x8*)&Ws[0][wn * 72 + wseg * 16 + 8] = wr[1];
  __syncthreads();

  for (int i = 0; i < 16; ++i) {
    const int cur = i & 1;
    if (i < 15) {  // issue next tile's global loads; WMMA overlaps them
      const int k0 = (i + 1) * 64;
#pragma unroll
      for (int j = 0; j < 4; ++j) xr[j] = *(const bf16x8*)(xg + k0 + j * 8);
      wr[0] = *(const bf16x8*)(wg + k0);
      wr[1] = *(const bf16x8*)(wg + k0 + 8);
    }
#pragma unroll
    for (int kk = 0; kk < 2; ++kk) {
      const bf16x16 a = frag_a(&Xs[cur][w * 16 * 72], 72, kk * 32);
#pragma unroll
      for (int t = 0; t < 4; ++t)
        acc[t] = wmma_bf16(a, frag_b(&Ws[cur][t * 16 * 72], 72, kk * 32), acc[t]);
    }
    if (i < 15) {
      const int nxt = cur ^ 1;
#pragma unroll
      for (int j = 0; j < 4; ++j) *(bf16x8*)&Xs[nxt][xrow * 72 + xh * 32 + j * 8] = xr[j];
      *(bf16x8*)&Ws[nxt][wn * 72 + wseg * 16] = wr[0];
      *(bf16x8*)&Ws[nxt][wn * 72 + wseg * 16 + 8] = wr[1];
    }
    __syncthreads();
  }
  // epilogue: + bias, store bf16
#pragma unroll
  for (int t = 0; t < 4; ++t) {
    const int cg = n0 + t * 16 + col;
    const float bv = bias[cg];
#pragma unroll
    for (int r = 0; r < 8; ++r) {
      const int rg = m0 + w * 16 + r + half * 8;
      O[(size_t)rg * 1024 + cg] = (bf16)(acc[t][r] + bv);
    }
  }
}

// ---------------------------------------------------------------------------
// Attention: per (b,h), 128-query block/WG; 32-key chunks, double-buffered
// K / V^T staging with register pipelining, ONE barrier per chunk.
// scores = exp((Q K^T)/8) * mask[key]; ctx = scores @ V; /(rowsum+1e-8).
// Q,K layout [B][L][H][64]; VT layout [B*H][64][L].
// ---------------------------------------------------------------------------
__global__ __launch_bounds__(256) void mha_attn(const bf16* __restrict__ Q,
                                                const bf16* __restrict__ K,
                                                const bf16* __restrict__ VT,
                                                const float* __restrict__ mask,
                                                float* __restrict__ out) {
  __shared__ __align__(16) bf16 Qs[128 * 72];      // [qrow][d]
  __shared__ __align__(16) bf16 Ks[2][32 * 72];    // [key][d] == B^T for QK^T
  __shared__ __align__(16) bf16 VTs[2][64 * 40];   // [d][key] == B^T for PV
  __shared__ __align__(16) bf16 Ps[8 * 16 * 40];   // per-wave P tile 16x32
  const int bh = blockIdx.y;
  const int b = bh >> 4, h = bh & 15;
  const int l0 = blockIdx.x * 128;
  const int tid = threadIdx.x;
  const int w = tid >> 5, lane = tid & 31;
  const int col = lane & 15, half = lane >> 4;

  // staging maps (8 bf16 per thread each)
  const int krow = tid >> 3, kseg = tid & 7;
  const int vd = tid >> 2, vseg = tid & 3;
  const bf16* kg = K + ((size_t)((b * 1024 + krow) * 16 + h)) * 64 + kseg * 8;
  const bf16* vg = VT + ((size_t)(bh * 64 + vd)) * 1024 + vseg * 8;

  // prologue: stage chunk 0 + Q block
  bf16x8 krg = *(const bf16x8*)kg;
  bf16x8 vrg = *(const bf16x8*)vg;
  *(bf16x8*)&Ks[0][krow * 72 + kseg * 8] = krg;
  *(bf16x8*)&VTs[0][vd * 40 + vseg * 8] = vrg;
  {
    const int row = tid >> 1, h2 = tid & 1;
    const bf16* g = Q + ((size_t)((b * 1024 + l0 + row) * 16 + h)) * 64 + h2 * 32;
#pragma unroll
    for (int i = 0; i < 4; ++i)
      *(bf16x8*)&Qs[row * 72 + h2 * 32 + i * 8] = *(const bf16x8*)(g + i * 8);
  }
  __syncthreads();

  // Q fragments are loop-invariant: hoist them
  const bf16x16 aq0 = frag_a(&Qs[w * 16 * 72], 72, 0);
  const bf16x16 aq1 = frag_a(&Qs[w * 16 * 72], 72, 32);

  f32x8 ctx[4];
#pragma unroll
  for (int t = 0; t < 4; ++t) { f32x8 z = {}; ctx[t] = z; }
  float rs[8];
#pragma unroll
  for (int r = 0; r < 8; ++r) rs[r] = 0.f;

  for (int ci = 0; ci < 32; ++ci) {
    const int cur = ci & 1;
    const int m0 = ci * 32;
    if (ci < 31) {  // prefetch next chunk into registers (overlaps WMMA)
      krg = *(const bf16x8*)(kg + (size_t)(ci + 1) * 1024);  // per-l stride = H*64
      vrg = *(const bf16x8*)(vg + (ci + 1) * 32);            // per-key stride = 1
    }
#pragma unroll
    for (int nt = 0; nt < 2; ++nt) {
      f32x8 s = {};
      s = wmma_bf16(aq0, frag_b(&Ks[cur][nt * 16 * 72], 72, 0), s);
      s = wmma_bf16(aq1, frag_b(&Ks[cur][nt * 16 * 72], 72, 32), s);
      const float mk = mask[b * 1024 + m0 + nt * 16 + col];
#pragma unroll
      for (int r = 0; r < 8; ++r) {
        const float p = __expf(s[r] * 0.125f) * mk;  // 1/sqrt(64)
        rs[r] += p;
        Ps[w * 640 + (r + half * 8) * 40 + nt * 16 + col] = (bf16)p;
      }
    }
    // P (16x32) x V (32x64): C-layout -> A-layout via per-wave LDS slab
    const bf16x16 ap = frag_a(&Ps[w * 640], 40, 0);
#pragma unroll
    for (int dt = 0; dt < 4; ++dt)
      ctx[dt] = wmma_bf16(ap, frag_b(&VTs[cur][dt * 16 * 40], 40, 0), ctx[dt]);
    if (ci < 31) {
      const int nxt = cur ^ 1;
      *(bf16x8*)&Ks[nxt][krow * 72 + kseg * 8] = krg;
      *(bf16x8*)&VTs[nxt][vd * 40 + vseg * 8] = vrg;
    }
    __syncthreads();
  }

  // row-sum reduce within each 16-lane half (xor masks <16 stay in-half)
#pragma unroll
  for (int r = 0; r < 8; ++r) {
#pragma unroll
    for (int m = 1; m <= 8; m <<= 1) rs[r] += __shfl_xor(rs[r], m, 32);
    rs[r] = 1.0f / (rs[r] + 1e-8f);
  }
#pragma unroll
  for (int dt = 0; dt < 4; ++dt) {
#pragma unroll
    for (int r = 0; r < 8; ++r) {
      const int l = l0 + w * 16 + r + half * 8;
      out[(size_t)(b * 1024 + l) * 1024 + h * 64 + dt * 16 + col] = ctx[dt][r] * rs[r];
    }
  }
}

// ---------------------------------------------------------------------------
// Launch: convert/transpose -> 3x GEMM (Q,K,V) -> V transpose -> attention
// ---------------------------------------------------------------------------
extern "C" void kernel_launch(void* const* d_in, const int* in_sizes, int n_in,
                              void* d_out, int out_size, void* d_ws, size_t ws_size,
                              hipStream_t stream) {
  const float* x    = (const float*)d_in[0];
  const float* mask = (const float*)d_in[1];
  const float* Wq   = (const float*)d_in[2];
  const float* bq   = (const float*)d_in[3];
  const float* Wk   = (const float*)d_in[4];
  const float* bk   = (const float*)d_in[5];
  const float* Wv   = (const float*)d_in[6];
  const float* bv   = (const float*)d_in[7];
  float* out = (float*)d_out;

  char* ws = (char*)d_ws;
  // workspace layout (bytes); vtb reuses the xb region (x dead after GEMMs)
  bf16* xb  = (bf16*)(ws + 0);          // 16 MiB
  bf16* wqt = (bf16*)(ws + 16777216);   // 2 MiB each, pre-transposed [n][k]
  bf16* wkt = (bf16*)(ws + 18874368);
  bf16* wvt = (bf16*)(ws + 20971520);
  bf16* qb  = (bf16*)(ws + 23068672);   // 16 MiB each
  bf16* kb  = (bf16*)(ws + 39845888);
  bf16* vb  = (bf16*)(ws + 56623104);
  bf16* vtb = (bf16*)(ws + 0);          // aliases xb (safe: stream-ordered)

  cvt_f32_bf16<<<8192, 256, 0, stream>>>(x, xb, (8 * 1024 * 1024) / 4);
  cvt_w_transpose<<<4096, 256, 0, stream>>>(Wq, wqt);
  cvt_w_transpose<<<4096, 256, 0, stream>>>(Wk, wkt);
  cvt_w_transpose<<<4096, 256, 0, stream>>>(Wv, wvt);

  dim3 gg(1024 / 64, 8192 / 128);
  gemm_qkv<<<gg, 256, 0, stream>>>(xb, wqt, bq, qb);
  gemm_qkv<<<gg, 256, 0, stream>>>(xb, wkt, bk, kb);
  gemm_qkv<<<gg, 256, 0, stream>>>(xb, wvt, bv, vb);

  dim3 gt(1024 / 64, 8 * 16);
  transpose_v<<<gt, 256, 0, stream>>>(vb, vtb);  // overwrites xb region

  dim3 ga(1024 / 128, 8 * 16);
  mha_attn<<<ga, 256, 0, stream>>>(qb, kb, vtb, mask, out);
}
